// SingleResolutionGraphModel_63428077027444
// MI455X (gfx1250) — matrix operand order repaired and checked by reading, not compiled, not run on hardware
//
#include <hip/hip_runtime.h>
#include <hip/hip_bf16.h>

// ---------------------------------------------------------------------------
// Problem constants (match reference)
// ---------------------------------------------------------------------------
#define NN   50000
#define EE   800000
#define FF   512
#define HH   2
#define CC   128
#define DD   256          // H*C
#define GG   8
#define NCLS 4

typedef __attribute__((ext_vector_type(16))) _Float16 v16h;
typedef __attribute__((ext_vector_type(8)))  float    v8f;

union Frag16 { v16h v; uint4 u[2]; };

// ---------------------------------------------------------------------------
// Helpers
// ---------------------------------------------------------------------------
__device__ __forceinline__ void atomicMaxF(float* addr, float val) {
    // IEEE trick: for val >= 0, int-max; for val < 0, uint-min.
    if (val >= 0.0f) atomicMax((int*)addr, __float_as_int(val));
    else             atomicMin((unsigned int*)addr, (unsigned int)__float_as_int(val));
}

__global__ void fill_f32_kernel(float* __restrict__ p, float v, int n) {
    int i = blockIdx.x * blockDim.x + threadIdx.x;
    if (i < n) p[i] = v;
}

__global__ void cvt_f32_to_f16_kernel(const float* __restrict__ in,
                                      _Float16* __restrict__ out, int n) {
    int i = blockIdx.x * blockDim.x + threadIdx.x;
    if (i < n) out[i] = (_Float16)in[i];
}

// W [K x Ncols] f32 (row-major)  ->  Wt [Ncols x K] f16 (row-major)
__global__ void wt_transpose_kernel(const float* __restrict__ W,
                                    _Float16* __restrict__ Wt,
                                    int K, int Ncols) {
    int i = blockIdx.x * blockDim.x + threadIdx.x;
    if (i >= K * Ncols) return;
    int k = i / Ncols;
    int n = i - k * Ncols;
    Wt[(size_t)n * K + k] = (_Float16)W[i];
}

// ---------------------------------------------------------------------------
// WMMA GEMM:  C[50000 x 256] = A[50000 x K](f16) * Bt^T + bias
//   Bt is [256 x K] f16 (pre-transposed weights, row n = output column n).
//   grid.x = 3125 (16-row stripes), block = 256 (8 waves).
//   Wave w computes column tiles {2w, 2w+1}; A fragment shared between both.
// ---------------------------------------------------------------------------
__global__ void __launch_bounds__(256)
gemm_f16_wmma_kernel(const _Float16* __restrict__ A,
                     const _Float16* __restrict__ Bt,
                     const float*    __restrict__ bias,
                     float*          __restrict__ C,
                     int K) {
    const int lane = threadIdx.x & 31;
    const int wave = threadIdx.x >> 5;
    const int hh   = lane >> 4;      // half-wave: 0 or 1
    const int l16  = lane & 15;

    const int row = blockIdx.x * 16 + l16;            // A row this lane feeds
    const _Float16* arow = A + (size_t)row * K;

    const int n_a = wave * 32 + l16;                  // first col tile column
    const int n_b = n_a + 16;                         // second col tile column
    const _Float16* brow0 = Bt + (size_t)n_a * K + hh * 8;
    const _Float16* brow1 = Bt + (size_t)n_b * K + hh * 8;

    v8f acc0 = {};
    v8f acc1 = {};

    for (int k0 = 0; k0 < K; k0 += 32) {
        Frag16 a, b0, b1;
        // A 16x32 f16 layout: lane(row, half) holds K = k0+hh*8+0..7 and k0+16+hh*8+0..7
        a.u[0] = *reinterpret_cast<const uint4*>(arow + k0 + hh * 8);
        a.u[1] = *reinterpret_cast<const uint4*>(arow + k0 + 16 + hh * 8);
        // B 32x16 f16 layout: lane(col, half) holds K = k0+hh*8+0..7 and k0+16+hh*8+0..7
        const uint4* bp0 = reinterpret_cast<const uint4*>(brow0 + k0);
        b0.u[0] = bp0[0]; b0.u[1] = *reinterpret_cast<const uint4*>(brow0 + k0 + 16);
        const uint4* bp1 = reinterpret_cast<const uint4*>(brow1 + k0);
        b1.u[0] = bp1[0]; b1.u[1] = *reinterpret_cast<const uint4*>(brow1 + k0 + 16);

        acc0 = __builtin_amdgcn_wmma_f32_16x16x32_f16(false, a.v, false, b0.v,
                                                      (short)0, acc0, false, false);
        acc1 = __builtin_amdgcn_wmma_f32_16x16x32_f16(false, a.v, false, b1.v,
                                                      (short)0, acc1, false, false);
    }

    const float bias0 = bias[n_a];
    const float bias1 = bias[n_b];
    // C/D layout: lane holds column N = l16, rows M = r + 8*hh (r = 0..7)
    const size_t base = ((size_t)blockIdx.x * 16 + 8 * hh) * DD;
#pragma unroll
    for (int r = 0; r < 8; ++r) {
        C[base + (size_t)r * DD + n_a] = acc0[r] + bias0;
        C[base + (size_t)r * DD + n_b] = acc1[r] + bias1;
    }
}

// ---------------------------------------------------------------------------
// Edge attention pipeline
// ---------------------------------------------------------------------------

// One wave per (edge, head): score = scale * dot(q[dst][h], k[src][h]) over 128 ch.
__global__ void __launch_bounds__(256)
edge_scores_kernel(const float* __restrict__ Q, const float* __restrict__ Kf,
                   const float* __restrict__ V,
                   const int* __restrict__ src, const int* __restrict__ dst,
                   float* __restrict__ sc, float scale) {
    const int eh   = blockIdx.x * 8 + (threadIdx.x >> 5);
    const int lane = threadIdx.x & 31;
    if (eh >= 2 * EE) return;
    const int e = eh >> 1;
    const int h = eh & 1;
    const int s = src[e];
    const int d = dst[e];

    const float4 qv = *reinterpret_cast<const float4*>(Q  + (size_t)d * DD + h * CC + lane * 4);
    const float4 kv = *reinterpret_cast<const float4*>(Kf + (size_t)s * DD + h * CC + lane * 4);
    // warm L2 for the value gather in the aggregation pass
    __builtin_prefetch(V + (size_t)s * DD + h * CC + lane * 4, 0, 1);

    float p = qv.x * kv.x + qv.y * kv.y + qv.z * kv.z + qv.w * kv.w;
#pragma unroll
    for (int off = 16; off >= 1; off >>= 1)
        p += __shfl_xor(p, off, 32);
    if (lane == 0) sc[eh] = p * scale;
}

// Segment max over destination nodes.
__global__ void edge_segmax_kernel(const float* __restrict__ sc,
                                   const int* __restrict__ dst,
                                   float* __restrict__ nmax) {
    int i = blockIdx.x * blockDim.x + threadIdx.x;
    if (i >= 2 * EE) return;
    int d = dst[i >> 1];
    atomicMaxF(&nmax[d * 2 + (i & 1)], sc[i]);
}

// ex = exp(score - max); accumulate denominator.
__global__ void edge_expsum_kernel(float* __restrict__ sc,
                                   const int* __restrict__ dst,
                                   const float* __restrict__ nmax,
                                   float* __restrict__ den) {
    int i = blockIdx.x * blockDim.x + threadIdx.x;
    if (i >= 2 * EE) return;
    int d  = dst[i >> 1];
    int h  = i & 1;
    float ex = __expf(sc[i] - nmax[d * 2 + h]);
    sc[i] = ex;
    atomicAdd(&den[d * 2 + h], ex);
}

// One block (256 thr) per edge: S[dst] += alpha * V[src]  (S already holds skip).
__global__ void __launch_bounds__(256)
edge_aggregate_kernel(const float* __restrict__ sc,
                      const float* __restrict__ den,
                      const float* __restrict__ V,
                      const int* __restrict__ src, const int* __restrict__ dst,
                      float* __restrict__ S) {
    const int e = blockIdx.x;
    const int c = threadIdx.x;          // 0..255
    const int h = c >> 7;
    const int s = src[e];
    const int d = dst[e];
    const float alpha = sc[e * 2 + h] / (den[d * 2 + h] + 1e-16f);
    atomicAdd(&S[(size_t)d * DD + c], alpha * V[(size_t)s * DD + c]);
}

__global__ void relu_kernel(float* __restrict__ p, int n) {
    int i = blockIdx.x * blockDim.x + threadIdx.x;
    if (i < n) p[i] = fmaxf(p[i], 0.0f);
}

// global_max_pool over batch assignment: one block per node, 256 channels.
__global__ void __launch_bounds__(256)
pool_max_kernel(const float* __restrict__ Hf, const int* __restrict__ batch,
                float* __restrict__ pool) {
    const int n = blockIdx.x;
    const int c = threadIdx.x;
    const int g = batch[n];
    atomicMaxF(&pool[g * DD + c], Hf[(size_t)n * DD + c]);
}

// Tiny MLP head: [8,256] @ [256,64] + b -> [8,64]; @ [64,4] + b -> [8,4].
__global__ void __launch_bounds__(512)
head_kernel(const float* __restrict__ pool,
            const float* __restrict__ wf1, const float* __restrict__ bf1,
            const float* __restrict__ wf2, const float* __restrict__ bf2,
            float* __restrict__ out) {
    __shared__ float hid[GG * 64];
    const int t = threadIdx.x;
    if (t < GG * 64) {
        const int g = t >> 6;
        const int j = t & 63;
        float acc = bf1[j];
        for (int k = 0; k < DD; ++k)
            acc += pool[g * DD + k] * wf1[k * 64 + j];
        hid[g * 64 + j] = acc;
    }
    __syncthreads();
    if (t < GG * NCLS) {
        const int g = t >> 2;
        const int j = t & 3;
        float acc = bf2[j];
        for (int k = 0; k < 64; ++k)
            acc += hid[g * 64 + k] * wf2[k * NCLS + j];
        out[g * NCLS + j] = acc;
    }
}

// ---------------------------------------------------------------------------
// Launch
// ---------------------------------------------------------------------------
extern "C" void kernel_launch(void* const* d_in, const int* in_sizes, int n_in,
                              void* d_out, int out_size, void* d_ws, size_t ws_size,
                              hipStream_t stream) {
    (void)in_sizes; (void)n_in; (void)out_size; (void)ws_size;

    const float* x    = (const float*)d_in[0];
    const int*   eidx = (const int*)d_in[1];
    const int*   src  = eidx;
    const int*   dst  = eidx + EE;
    const int*   batch = (const int*)d_in[2];

    const float* Wl[8] = { (const float*)d_in[3],  (const float*)d_in[5],
                           (const float*)d_in[7],  (const float*)d_in[9],
                           (const float*)d_in[11], (const float*)d_in[13],
                           (const float*)d_in[15], (const float*)d_in[17] };
    const float* Bl[8] = { (const float*)d_in[4],  (const float*)d_in[6],
                           (const float*)d_in[8],  (const float*)d_in[10],
                           (const float*)d_in[12], (const float*)d_in[14],
                           (const float*)d_in[16], (const float*)d_in[18] };
    const float* wf1 = (const float*)d_in[19];
    const float* bf1 = (const float*)d_in[20];
    const float* wf2 = (const float*)d_in[21];
    const float* bf2 = (const float*)d_in[22];
    float* out = (float*)d_out;

    // ---- workspace partition ------------------------------------------------
    char* p = (char*)d_ws;
    auto take = [&](size_t bytes) {
        char* r = p;
        p += (bytes + 255) & ~(size_t)255;
        return r;
    };
    _Float16* X16  = (_Float16*)take((size_t)NN * FF * sizeof(_Float16)); // also h1-f16
    _Float16* WT[8];
    for (int i = 0; i < 4; ++i) WT[i]     = (_Float16*)take((size_t)DD * FF * sizeof(_Float16));
    for (int i = 0; i < 4; ++i) WT[4 + i] = (_Float16*)take((size_t)DD * DD * sizeof(_Float16));
    float* Qb  = (float*)take((size_t)NN * DD * sizeof(float));
    float* Kb  = (float*)take((size_t)NN * DD * sizeof(float));
    float* Vb  = (float*)take((size_t)NN * DD * sizeof(float));
    float* Sb  = (float*)take((size_t)NN * DD * sizeof(float));   // skip + agg + relu (h)
    float* SC  = (float*)take((size_t)EE * HH * sizeof(float));
    float* NM  = (float*)take((size_t)NN * HH * sizeof(float));
    float* DEN = (float*)take((size_t)NN * HH * sizeof(float));
    float* POOL = (float*)take((size_t)GG * DD * sizeof(float));

    const float scale = 0.088388347648318447f;   // 1/sqrt(128)
    const float NEGINF = -__builtin_inff();
    const int BLK = 256;

    // ---- weight prep: f32 -> f16 transposed ---------------------------------
    for (int i = 0; i < 4; ++i) {
        int tot = FF * DD;
        wt_transpose_kernel<<<(tot + BLK - 1) / BLK, BLK, 0, stream>>>(Wl[i], WT[i], FF, DD);
    }
    for (int i = 0; i < 4; ++i) {
        int tot = DD * DD;
        wt_transpose_kernel<<<(tot + BLK - 1) / BLK, BLK, 0, stream>>>(Wl[4 + i], WT[4 + i], DD, DD);
    }

    // ---- input x -> f16 -----------------------------------------------------
    {
        int tot = NN * FF;
        cvt_f32_to_f16_kernel<<<(tot + BLK - 1) / BLK, BLK, 0, stream>>>(x, X16, tot);
    }

    const int gemmBlocks = NN / 16;                       // 3125, exact
    const int ehBlocks   = (2 * EE + 7) / 8;              // 1 wave per (edge,head)
    const int ehThreads  = (2 * EE + BLK - 1) / BLK;

    for (int layer = 0; layer < 2; ++layer) {
        const int K = (layer == 0) ? FF : DD;
        const int w0 = layer * 4;

        // Q/K/V/skip GEMMs (WMMA)
        gemm_f16_wmma_kernel<<<gemmBlocks, BLK, 0, stream>>>(X16, WT[w0 + 0], Bl[w0 + 0], Qb, K);
        gemm_f16_wmma_kernel<<<gemmBlocks, BLK, 0, stream>>>(X16, WT[w0 + 1], Bl[w0 + 1], Kb, K);
        gemm_f16_wmma_kernel<<<gemmBlocks, BLK, 0, stream>>>(X16, WT[w0 + 2], Bl[w0 + 2], Vb, K);
        gemm_f16_wmma_kernel<<<gemmBlocks, BLK, 0, stream>>>(X16, WT[w0 + 3], Bl[w0 + 3], Sb, K);

        // segment softmax state
        fill_f32_kernel<<<(NN * HH + BLK - 1) / BLK, BLK, 0, stream>>>(NM, NEGINF, NN * HH);
        fill_f32_kernel<<<(NN * HH + BLK - 1) / BLK, BLK, 0, stream>>>(DEN, 0.0f, NN * HH);

        edge_scores_kernel<<<ehBlocks, BLK, 0, stream>>>(Qb, Kb, Vb, src, dst, SC, scale);
        edge_segmax_kernel<<<ehThreads, BLK, 0, stream>>>(SC, dst, NM);
        edge_expsum_kernel<<<ehThreads, BLK, 0, stream>>>(SC, dst, NM, DEN);
        edge_aggregate_kernel<<<EE, BLK, 0, stream>>>(SC, DEN, Vb, src, dst, Sb);

        relu_kernel<<<(NN * DD + BLK - 1) / BLK, BLK, 0, stream>>>(Sb, NN * DD);

        if (layer == 0) {
            // h1 -> f16 (reuse X16) for layer-2 GEMMs
            int tot = NN * DD;
            cvt_f32_to_f16_kernel<<<(tot + BLK - 1) / BLK, BLK, 0, stream>>>(Sb, X16, tot);
        }
    }

    // ---- global max pool + MLP head ----------------------------------------
    fill_f32_kernel<<<(GG * DD + BLK - 1) / BLK, BLK, 0, stream>>>(POOL, NEGINF, GG * DD);
    pool_max_kernel<<<NN, BLK, 0, stream>>>(Sb, batch, POOL);
    head_kernel<<<1, 512, 0, stream>>>(POOL, wf1, bf1, wf2, bf2, out);
}
